// DefVoxelSetAbstraction_12403865551663
// MI455X (gfx1250) — compile-verified, matrix operand order checked by reading, not run on hardware
//
#include <hip/hip_runtime.h>
#include <hip/hip_bf16.h>
#include <math.h>

typedef __attribute__((ext_vector_type(16))) _Float16 v16h;
typedef __attribute__((ext_vector_type(8)))  float    v8f;
typedef __attribute__((ext_vector_type(4)))  int      v4i;

#if defined(__gfx1250__) && __has_builtin(__builtin_amdgcn_global_load_async_to_lds_b128)
#define USE_ASYNC_LDS 1
#else
#define USE_ASYNC_LDS 0
#endif

__device__ __forceinline__ void async_wait_all() {
#if USE_ASYNC_LDS
#if __has_builtin(__builtin_amdgcn_s_wait_asynccnt)
  __builtin_amdgcn_s_wait_asynccnt(0);
#else
  asm volatile("s_wait_asynccnt 0" ::: "memory");
#endif
#endif
}

#if USE_ASYNC_LDS
// async 16B global->LDS copy; builtin takes (global v4i*, local v4i*, imm, imm)
__device__ __forceinline__ void async_copy16(const _Float16* g, _Float16* l) {
  void* gp = (void*)g;  // shed const
  __builtin_amdgcn_global_load_async_to_lds_b128(
      (__attribute__((address_space(1))) v4i*)gp,
      (__attribute__((address_space(3))) v4i*)l,
      0, 0);
}
#endif

// -------- fragment loader: 16 f16 for the CDNA5 16-bit A/B WMMA layout --------
// Lane supplies base pointer at (row, kbase + 8*hi); elements 0..7 = K..K+7,
// elements 8..15 = K+16..K+23 (matches ISA 16-bit 16x32 A layout, mirrored B).
union FragU { v16h v; float4 f[2]; };

__device__ __forceinline__ v16h load_frag16(const _Float16* p) {
  FragU u;
  u.f[0] = *reinterpret_cast<const float4*>(p);
  u.f[1] = *reinterpret_cast<const float4*>(p + 16);
  return u.v;
}

// ---------------- max over the 10 samples of x: [1,6,N,10] -> F0 [N,32] f16 ---
__global__ void maxpool_x_kernel(const float* __restrict__ x,
                                 _Float16* __restrict__ F0, int N) {
  int n = blockIdx.x * blockDim.x + threadIdx.x;
  if (n >= N) return;
#pragma unroll
  for (int c = 0; c < 6; ++c) {
    const float* px = x + ((size_t)c * N + n) * 10;
    float m = px[0];
#pragma unroll
    for (int j = 1; j < 10; ++j) m = fmaxf(m, px[j]);
    F0[(size_t)n * 32 + c] = (_Float16)m;
  }
#pragma unroll
  for (int c = 6; c < 32; ++c) F0[(size_t)n * 32 + c] = (_Float16)0.f;
}

// ---------------- weight prep: W[O,2C] fp32 -> Wd=Wa, Wf=Wb-Wa (f16, [O,C]) ---
__global__ void prep_w_kernel(const float* __restrict__ W,
                              _Float16* __restrict__ Wd, _Float16* __restrict__ Wf,
                              int O, int C) {
  int i = blockIdx.x * blockDim.x + threadIdx.x;
  if (i >= O * C) return;
  int o = i / C, c = i % C;
  float wa = W[(size_t)o * 2 * C + c];
  float wb = W[(size_t)o * 2 * C + C + c];
  Wd[i] = (_Float16)wa;
  Wf[i] = (_Float16)(wb - wa);
}

// Layer 1: W1[64,9] over e=[nbp-cp; cp; nbc-cc].
// Neighbor f6=[p;c]: Wd'=[W1a|W1c]; center: Wf'=[W1b-W1a | -W1c]; pad C to 32.
__global__ void prep_w1_kernel(const float* __restrict__ W1,
                               _Float16* __restrict__ Wd, _Float16* __restrict__ Wf) {
  int i = blockIdx.x * blockDim.x + threadIdx.x;
  if (i >= 64 * 32) return;
  int o = i >> 5, c = i & 31;
  float wd = 0.f, wf = 0.f;
  if (c < 3)      { wd = W1[o * 9 + c];     wf = W1[o * 9 + 3 + c] - W1[o * 9 + c]; }
  else if (c < 6) { wd = W1[o * 9 + 3 + c]; wf = -W1[o * 9 + 3 + c]; } // 6+(c-3)=3+c
  Wd[i] = (_Float16)wd;
  Wf[i] = (_Float16)wf;
}

__global__ void prep_w6_kernel(const float* __restrict__ W6, _Float16* __restrict__ W6h) {
  int i = blockIdx.x * blockDim.x + threadIdx.x;
  if (i < 32 * 512) W6h[i] = (_Float16)W6[i];
}

// ---------------- KNN (top-10 by smallest squared distance, self included) ----
__global__ void knn_kernel(const _Float16* __restrict__ F, int ldF, int coff,
                           int* __restrict__ idx, int N) {
  __shared__ float sx[256], sy[256], sz[256];
  int n = blockIdx.x * blockDim.x + threadIdx.x;
  const _Float16* fp = F + (size_t)n * ldF + coff;
  float px = (float)fp[0], py = (float)fp[1], pz = (float)fp[2];
  float bd[10]; int bi[10];
#pragma unroll
  for (int q = 0; q < 10; ++q) { bd[q] = -3.0e38f; bi[q] = 0; }
  for (int t0 = 0; t0 < N; t0 += 256) {
    const _Float16* fc = F + (size_t)(t0 + threadIdx.x) * ldF + coff;
    sx[threadIdx.x] = (float)fc[0];
    sy[threadIdx.x] = (float)fc[1];
    sz[threadIdx.x] = (float)fc[2];
    __syncthreads();
    for (int jj = 0; jj < 256; ++jj) {
      float dx = px - sx[jj], dy = py - sy[jj], dz = pz - sz[jj];
      float nd = -(dx * dx + dy * dy + dz * dz);     // neg sq dist, maximize
      if (nd > bd[9]) {
        bd[9] = nd; bi[9] = t0 + jj;
#pragma unroll
        for (int q = 9; q > 0; --q) {
          if (bd[q] > bd[q - 1]) {
            float td = bd[q]; bd[q] = bd[q - 1]; bd[q - 1] = td;
            int   ti = bi[q]; bi[q] = bi[q - 1]; bi[q - 1] = ti;
          }
        }
      }
    }
    __syncthreads();
  }
#pragma unroll
  for (int q = 0; q < 10; ++q) idx[(size_t)n * 10 + q] = bi[q];
}

// ---------------- fused edge-conv: async gather -> LDS -> WMMA -> max_k -------
// One workgroup owns a 16-point tile and stages its 160 neighbor rows + 16
// center rows into LDS ONCE (async global->LDS on gfx1250), shared by all
// blockDim.y waves; wave y computes output channels [y*16, y*16+16).
template <int NK>
__global__ void edgeconv_wmma_kernel(const _Float16* __restrict__ F, int ldF, int coff,
                                     const int* __restrict__ idx,
                                     const _Float16* __restrict__ Wd,
                                     const _Float16* __restrict__ Wf,
                                     const float* __restrict__ s,
                                     const float* __restrict__ b,
                                     _Float16* __restrict__ Fout, int ldO, int ocoff,
                                     int N) {
  constexpr int CH  = NK * 32;
  constexpr int KNB = 10;
  constexpr int ROWS = 16 * KNB + 16;     // 160 neighbor rows (j-major) + 16 center
  constexpr int LDA  = CH + 8;            // +16B pad to stagger LDS banks
  __shared__ _Float16 sA[ROWS * LDA];
  __shared__ int sIdx[16 * KNB];

  const int m0   = blockIdx.x << 4;
  const int lane = threadIdx.x;
  const int r16  = lane & 15;
  const int hi   = lane >> 4;
  const int tid  = threadIdx.y * 32 + lane;
  const int nthr = blockDim.y * 32;

  // stage neighbor indices (row j*16+pi holds neighbor j of point m0+pi)
  for (int i = tid; i < 16 * KNB; i += nthr) {
    int j = i >> 4, pi = i & 15;
    sIdx[i] = idx[(size_t)(m0 + pi) * KNB + j];
  }
  __syncthreads();

  // cooperative gather of ROWS x CH f16 into LDS in 16B chunks
  constexpr int cpr = CH / 8;             // 16B chunks per row
  for (int ch = tid; ch < ROWS * cpr; ch += nthr) {
    int r  = ch / cpr;
    int cc = (ch - r * cpr) * 8;
    int src = (r < 16 * KNB) ? sIdx[r] : (m0 + (r - 16 * KNB));
    const _Float16* g = F + (size_t)src * ldF + coff + cc;
    _Float16* l = &sA[r * LDA + cc];
#if USE_ASYNC_LDS
    async_copy16(g, l);
#else
    *reinterpret_cast<float4*>(l) = *reinterpret_cast<const float4*>(g);
#endif
  }
  async_wait_all();
  __syncthreads();

  const int o = (threadIdx.y << 4) + r16;   // B col / D col (output channel)

  // Preload Wd B-fragments (reused k=10 times)
  v16h bw[NK];
#pragma unroll
  for (int t = 0; t < NK; ++t)
    bw[t] = load_frag16(Wd + (size_t)o * CH + t * 32 + 8 * hi);

  v8f maxv;
#pragma unroll
  for (int r = 0; r < 8; ++r) maxv[r] = -3.0e38f;

  for (int j = 0; j < KNB; ++j) {
    const _Float16* arow = &sA[(j * 16 + r16) * LDA + 8 * hi];
    v8f acc = {};
#pragma unroll
    for (int t = 0; t < NK; ++t) {
      v16h a = load_frag16(arow + t * 32);
      acc = __builtin_amdgcn_wmma_f32_16x16x32_f16(false, a, false, bw[t],
                                                   (short)0, acc, false, false);
    }
#pragma unroll
    for (int r = 0; r < 8; ++r) maxv[r] = fmaxf(maxv[r], acc[r]);
  }

  // Center contribution: (Wb - Wa) * ctr, neighbor-independent
  {
    const _Float16* arow = &sA[(16 * KNB + r16) * LDA + 8 * hi];
    v8f acc = {};
#pragma unroll
    for (int t = 0; t < NK; ++t) {
      v16h a  = load_frag16(arow + t * 32);
      v16h bf = load_frag16(Wf + (size_t)o * CH + t * 32 + 8 * hi);
      acc = __builtin_amdgcn_wmma_f32_16x16x32_f16(false, a, false, bf,
                                                   (short)0, acc, false, false);
    }
#pragma unroll
    for (int r = 0; r < 8; ++r) maxv[r] += acc[r];
  }

  float sv = s[o], bv = b[o];
#pragma unroll
  for (int r = 0; r < 8; ++r) {
    float y = sv * maxv[r] + bv;
    y = (y >= 0.f) ? y : 0.2f * y;
    int m = m0 + r + 8 * hi;              // D layout: lanes16-31 hold M = r+8
    Fout[(size_t)m * ldO + ocoff + o] = (_Float16)y;
  }
}

// ---------------- final 512 -> 32 GEMM + BN + LReLU, fp32 out [N,32] ----------
__global__ void final_gemm_wmma_kernel(const _Float16* __restrict__ F,   // [N,512]
                                       const _Float16* __restrict__ W6h, // [32,512]
                                       const float* __restrict__ s,
                                       const float* __restrict__ b,
                                       float* __restrict__ out, int N) {
  const int C = 512, NK = 16;
  int wid    = blockIdx.x * blockDim.y + threadIdx.y;
  int mtiles = N >> 4;
  int mtile  = wid % mtiles;
  int ntile  = wid / mtiles;   // 0..1
  int lane = threadIdx.x;
  int r16  = lane & 15;
  int hi   = lane >> 4;
  int m0   = mtile << 4;
  const _Float16* arow = F   + (size_t)(m0 + r16) * C + 8 * hi;
  const _Float16* brow = W6h + (size_t)((ntile << 4) + r16) * C + 8 * hi;
  v8f acc = {};
#pragma unroll
  for (int t = 0; t < NK; ++t) {
    v16h a  = load_frag16(arow + t * 32);
    v16h bb = load_frag16(brow + t * 32);
    acc = __builtin_amdgcn_wmma_f32_16x16x32_f16(false, a, false, bb,
                                                 (short)0, acc, false, false);
  }
  int o = (ntile << 4) + r16;
  float sv = s[o], bv = b[o];
#pragma unroll
  for (int r = 0; r < 8; ++r) {
    float y = sv * acc[r] + bv;
    y = (y >= 0.f) ? y : 0.2f * y;
    out[(size_t)(m0 + r + 8 * hi) * 32 + o] = y;
  }
}

// -----------------------------------------------------------------------------
extern "C" void kernel_launch(void* const* d_in, const int* in_sizes, int n_in,
                              void* d_out, int out_size, void* d_ws, size_t ws_size,
                              hipStream_t stream) {
  const float* x  = (const float*)d_in[0];
  const float* W1 = (const float*)d_in[1];
  const float* s1 = (const float*)d_in[2];
  const float* b1 = (const float*)d_in[3];
  const float* W2 = (const float*)d_in[4];
  const float* s2 = (const float*)d_in[5];
  const float* b2 = (const float*)d_in[6];
  const float* W3 = (const float*)d_in[7];
  const float* s3 = (const float*)d_in[8];
  const float* b3 = (const float*)d_in[9];
  const float* W4 = (const float*)d_in[10];
  const float* s4 = (const float*)d_in[11];
  const float* b4 = (const float*)d_in[12];
  const float* W6 = (const float*)d_in[13];
  const float* s6 = (const float*)d_in[14];
  const float* b6 = (const float*)d_in[15];

  const int N = in_sizes[0] / 60;   // [1,6,N,10]

  // workspace carve-out (256B aligned)
  size_t off = 0;
  auto carve = [&](size_t bytes) {
    size_t o = off;
    off = (off + bytes + 255) & ~(size_t)255;
    return (char*)d_ws + o;
  };
  _Float16* F0   = (_Float16*)carve((size_t)N * 32 * 2);
  _Float16* Fcat = (_Float16*)carve((size_t)N * 512 * 2);
  int*      idxb = (int*)     carve((size_t)N * 10 * 4);
  _Float16* Wd1  = (_Float16*)carve(64  * 32  * 2);
  _Float16* Wf1  = (_Float16*)carve(64  * 32  * 2);
  _Float16* Wd2  = (_Float16*)carve(64  * 64  * 2);
  _Float16* Wf2  = (_Float16*)carve(64  * 64  * 2);
  _Float16* Wd3  = (_Float16*)carve(128 * 64  * 2);
  _Float16* Wf3  = (_Float16*)carve(128 * 64  * 2);
  _Float16* Wd4  = (_Float16*)carve(256 * 128 * 2);
  _Float16* Wf4  = (_Float16*)carve(256 * 128 * 2);
  _Float16* W6h  = (_Float16*)carve(32  * 512 * 2);

  // ---- prep ----
  maxpool_x_kernel<<<(N + 255) / 256, 256, 0, stream>>>(x, F0, N);
  prep_w1_kernel<<<(64 * 32 + 255) / 256, 256, 0, stream>>>(W1, Wd1, Wf1);
  prep_w_kernel <<<(64 * 64   + 255) / 256, 256, 0, stream>>>(W2, Wd2, Wf2, 64, 64);
  prep_w_kernel <<<(128 * 64  + 255) / 256, 256, 0, stream>>>(W3, Wd3, Wf3, 128, 64);
  prep_w_kernel <<<(256 * 128 + 255) / 256, 256, 0, stream>>>(W4, Wd4, Wf4, 256, 128);
  prep_w6_kernel<<<(32 * 512 + 255) / 256, 256, 0, stream>>>(W6, W6h);

  const int mtiles = N >> 4;

  // layer 1: F0 (C=32 padded, 6 real) -> Fcat[:, 0:64)
  knn_kernel<<<N / 256, 256, 0, stream>>>(F0, 32, 0, idxb, N);
  edgeconv_wmma_kernel<1><<<mtiles, dim3(32, 4), 0, stream>>>(
      F0, 32, 0, idxb, Wd1, Wf1, s1, b1, Fcat, 512, 0, N);

  // layer 2: xw (cols 0..63) -> Fcat[:, 64:128)
  knn_kernel<<<N / 256, 256, 0, stream>>>(Fcat, 512, 0, idxb, N);
  edgeconv_wmma_kernel<2><<<mtiles, dim3(32, 4), 0, stream>>>(
      Fcat, 512, 0, idxb, Wd2, Wf2, s2, b2, Fcat, 512, 64, N);

  // layer 3: x2 (cols 64..127) -> Fcat[:, 128:256)
  knn_kernel<<<N / 256, 256, 0, stream>>>(Fcat, 512, 64, idxb, N);
  edgeconv_wmma_kernel<2><<<mtiles, dim3(32, 8), 0, stream>>>(
      Fcat, 512, 64, idxb, Wd3, Wf3, s3, b3, Fcat, 512, 128, N);

  // layer 4: x3 (cols 128..255) -> Fcat[:, 256:512)
  knn_kernel<<<N / 256, 256, 0, stream>>>(Fcat, 512, 128, idxb, N);
  edgeconv_wmma_kernel<4><<<mtiles, dim3(32, 16), 0, stream>>>(
      Fcat, 512, 128, idxb, Wd4, Wf4, s4, b4, Fcat, 512, 256, N);

  // final: Fcat [N,512] x W6h^T -> out [N,32] fp32
  final_gemm_wmma_kernel<<<dim3((unsigned)(mtiles * 2 / 4)), dim3(32, 4), 0, stream>>>(
      Fcat, W6h, s6, b6, (float*)d_out, N);
}
